// Attention_68109591380231
// MI455X (gfx1250) — matrix operand order repaired and checked by reading, not compiled, not run on hardware
//
#include <hip/hip_runtime.h>
#include <hip/hip_bf16.h>
#include <math.h>

typedef __attribute__((ext_vector_type(16))) __bf16 v16bf;
typedef __attribute__((ext_vector_type(8)))  __bf16 v8bf;
typedef __attribute__((ext_vector_type(8)))  float  v8f;
typedef __attribute__((ext_vector_type(4)))  unsigned int u32x4;
typedef __attribute__((ext_vector_type(8)))  unsigned int u32x8;

#define B_    2
#define S_    2048
#define HS_   2048
#define H_    16
#define HKV_  4
#define D_    128
#define NQKV_ 3072      /* H*D + 2*HKV*D */
#define NEGM_ (-1e9f)
#define EPS_  1e-6f

// ---------------------------------------------------------------------------
// f32 -> bf16 elementwise convert
// ---------------------------------------------------------------------------
__global__ void __launch_bounds__(256)
cvt_f32_bf16(const float* __restrict__ in, __bf16* __restrict__ out, int n) {
    int i = blockIdx.x * blockDim.x + threadIdx.x;
    if (i < n) out[i] = (__bf16)in[i];
}

// ---------------------------------------------------------------------------
// TDM: issue tensor_load_to_lds for a 2D bf16 tile (rows x cols) from a
// row-major [tensor_d1, tensor_d0] matrix. All args wave-uniform -> SGPRs.
// D# layout per CDNA5 ISA 8.3/8.4 (group0: 4 SGPRs, group1: 8 SGPRs).
// ---------------------------------------------------------------------------
__device__ __forceinline__ void tdm_load_2d_bf16(
    unsigned lds_addr, const __bf16* gptr,
    unsigned tensor_d0, unsigned tensor_d1,
    unsigned long long stride0_elems,
    unsigned tile_d0, unsigned tile_d1)
{
    unsigned long long ga = (unsigned long long)(uintptr_t)gptr;
    u32x4 g0;
    g0[0] = 1u;                                   // count=1, user-mode D#
    g0[1] = lds_addr;                             // LDS byte address
    g0[2] = (unsigned)(ga & 0xFFFFFFFFu);         // global_addr[31:0]
    g0[3] = (unsigned)((ga >> 32) & 0x1FFFFFFu)   // global_addr[56:32]
          | (2u << 30);                           // type = 2 ("image")
    u32x8 g1;
    g1[0] = (1u << 16);                           // wg_mask=0, data_size=1 (2B)
    g1[1] = (tensor_d0 & 0xFFFFu) << 16;          // tensor_dim0[15:0]
    g1[2] = (tensor_d0 >> 16) | ((tensor_d1 & 0xFFFFu) << 16);
    g1[3] = (tensor_d1 >> 16) | (tile_d0 << 16);  // tile_dim0
    g1[4] = tile_d1;                              // tile_dim1 (tile_dim2 = 0)
    g1[5] = (unsigned)(stride0_elems & 0xFFFFFFFFu);
    g1[6] = (unsigned)((stride0_elems >> 32) & 0xFFFFu);   // stride1 = 0
    g1[7] = 0u;
    asm volatile("tensor_load_to_lds %0, %1" :: "s"(g0), "s"(g1) : "memory");
}

// ---------------------------------------------------------------------------
// bf16 GEMM with TDM-staged, double-buffered LDS tiles:
//   C[M,N] (f32) = A[M,K] (bf16 row-major) * B[N,K]^T
// Block = 256 threads = 8 waves (2 M x 4 N); wave tile 32x64; block 64x256.
// Wave 0 DMAs the next 64x32 A-slab and 256x32 B-slab while all waves run
// WMMAs out of the current slab. 8 v_wmma per wave per K-step.
// ---------------------------------------------------------------------------
__global__ void __launch_bounds__(256)
gemm_bf16_f32(const __bf16* __restrict__ A, const __bf16* __restrict__ Bm,
              float* __restrict__ C, int M, int N, int K) {
    const int lane = threadIdx.x & 31;
    const int w    = threadIdx.x >> 5;      // 0..7
    const int wm   = w & 1;                 // 2 waves in M
    const int wn   = w >> 1;                // 4 waves in N
    const int r    = lane & 15;
    const int hh   = lane >> 4;
    const int mblk = blockIdx.y * 64;
    const int nblk = blockIdx.x * 256;

    __shared__ __align__(32) __bf16 ldsA[2][64 * 32];    // 8 KB
    __shared__ __align__(32) __bf16 ldsB[2][256 * 32];   // 32 KB

    v8f acc[2][4] = {};

    // prologue: stage K-slab 0
    if (w == 0) {
        tdm_load_2d_bf16((unsigned)(uintptr_t)&ldsA[0][0],
                         A + (size_t)mblk * K, (unsigned)K, (unsigned)M,
                         (unsigned long long)K, 32u, 64u);
        tdm_load_2d_bf16((unsigned)(uintptr_t)&ldsB[0][0],
                         Bm + (size_t)nblk * K, (unsigned)K, (unsigned)N,
                         (unsigned long long)K, 32u, 256u);
        __builtin_amdgcn_s_wait_tensorcnt(0);
    }
    __syncthreads();

    int p = 0;
    for (int kk = 0; kk < K; kk += 32) {
        // prefetch next slab into the other buffer (overlaps with WMMAs)
        if (kk + 32 < K && w == 0) {
            tdm_load_2d_bf16((unsigned)(uintptr_t)&ldsA[p ^ 1][0],
                             A + (size_t)mblk * K + kk + 32,
                             (unsigned)K, (unsigned)M,
                             (unsigned long long)K, 32u, 64u);
            tdm_load_2d_bf16((unsigned)(uintptr_t)&ldsB[p ^ 1][0],
                             Bm + (size_t)nblk * K + kk + 32,
                             (unsigned)K, (unsigned)N,
                             (unsigned long long)K, 32u, 256u);
        }
        // A fragments for this wave's two 16-row subtiles
        v16bf a[2];
#pragma unroll
        for (int mt = 0; mt < 2; ++mt) {
            const __bf16* ap = &ldsA[p][(wm * 32 + mt * 16 + r) * 32 + hh * 16];
            v8bf lo = *(const v8bf*)ap;
            v8bf hi = *(const v8bf*)(ap + 8);
#pragma unroll
            for (int j = 0; j < 8; ++j) { a[mt][j] = lo[j]; a[mt][8 + j] = hi[j]; }
        }
#pragma unroll
        for (int t = 0; t < 4; ++t) {
            const __bf16* bp = &ldsB[p][(wn * 64 + t * 16 + r) * 32 + hh * 16];
            v16bf bfrag = *(const v16bf*)bp;
            acc[0][t] = __builtin_amdgcn_wmma_f32_16x16x32_bf16(
                false, a[0], false, bfrag, (short)0, acc[0][t], false, false);
            acc[1][t] = __builtin_amdgcn_wmma_f32_16x16x32_bf16(
                false, a[1], false, bfrag, (short)0, acc[1][t], false, false);
        }
        if (w == 0) __builtin_amdgcn_s_wait_tensorcnt(0);
        __syncthreads();
        p ^= 1;
    }

#pragma unroll
    for (int mt = 0; mt < 2; ++mt)
#pragma unroll
        for (int t = 0; t < 4; ++t)
#pragma unroll
            for (int v = 0; v < 8; ++v)
                C[(size_t)(mblk + wm * 32 + mt * 16 + v + 8 * hh) * N
                  + nblk + wn * 64 + t * 16 + r] = acc[mt][t][v];
}

// ---------------------------------------------------------------------------
// Per-head RMS norm + RoPE; splits fused qkv f32 buffer into:
//   qh [B,H,S,D]  bf16,  kh [B,HKV,S,D] bf16,  vt [B,HKV,D,S] bf16 (transposed)
// ---------------------------------------------------------------------------
__global__ void __launch_bounds__(128)
qk_norm_rope(const float* __restrict__ qkv,
             const float* __restrict__ cosT, const float* __restrict__ sinT,
             const float* __restrict__ qw, const float* __restrict__ kw,
             __bf16* __restrict__ qh, __bf16* __restrict__ kh,
             __bf16* __restrict__ vt) {
    const int tid  = threadIdx.x;     // d index 0..127
    const int slot = blockIdx.x;      // 0..23
    const int s    = blockIdx.y;
    const int b    = blockIdx.z;

    __shared__ float red[4];
    __shared__ float nv[D_];

    float x = qkv[(size_t)(b * S_ + s) * NQKV_ + slot * D_ + tid];

    if (slot < H_ + HKV_) {
        float ss = x * x;
#pragma unroll
        for (int m = 16; m >= 1; m >>= 1) ss += __shfl_xor(ss, m, 32);
        if ((tid & 31) == 0) red[tid >> 5] = ss;
        __syncthreads();
        float tot = red[0] + red[1] + red[2] + red[3];
        float n = x * rsqrtf(tot * (1.0f / D_) + EPS_);
        n *= (slot < H_) ? qw[tid] : kw[tid];
        nv[tid] = n;
        __syncthreads();
        float part = (tid < 64) ? -nv[tid + 64] : nv[tid - 64];
        float o = n * cosT[s * D_ + tid] + part * sinT[s * D_ + tid];
        if (slot < H_) {
            qh[((size_t)(b * H_ + slot) * S_ + s) * D_ + tid] = (__bf16)o;
        } else {
            int hk = slot - H_;
            kh[((size_t)(b * HKV_ + hk) * S_ + s) * D_ + tid] = (__bf16)o;
        }
    } else {
        int hv = slot - H_ - HKV_;
        vt[((size_t)(b * HKV_ + hv) * D_ + tid) * S_ + s] = (__bf16)x;
    }
}

// ---------------------------------------------------------------------------
// Causal flash attention (GQA). Block = 128 threads = 4 waves; each wave owns
// one 16-query tile, streams keys in 32-wide chunks with online softmax.
// Per chunk: 8 WMMA (Q*K^T) + 8 WMMA (P*V); P goes C-layout -> LDS -> A-layout.
// ---------------------------------------------------------------------------
__global__ void __launch_bounds__(128)
flash_attn(const __bf16* __restrict__ qh, const __bf16* __restrict__ kh,
           const __bf16* __restrict__ vt, __bf16* __restrict__ aout) {
    const int lane = threadIdx.x & 31;
    const int w    = threadIdx.x >> 5;          // 0..3
    const int r    = lane & 15;
    const int hh   = lane >> 4;
    const int qt   = blockIdx.x * 4 + w;        // query tile (16 rows)
    const int h    = blockIdx.y;
    const int b    = blockIdx.z;
    const int hk   = h >> 2;                    // H/HKV = 4
    const float scale = 0.08838834764831845f;   // D^-0.5

    __shared__ __align__(32) __bf16 plds[4][16 * 32];
    __bf16* pl = plds[w];

    const __bf16* qbase = qh + ((size_t)(b * H_ + h) * S_ + qt * 16) * D_;
    const __bf16* kbase = kh + ((size_t)(b * HKV_ + hk) * S_) * D_;
    const __bf16* vbase = vt + ((size_t)(b * HKV_ + hk) * D_) * S_;

    v16bf qa[4];
#pragma unroll
    for (int kk = 0; kk < 4; ++kk) {
        const __bf16* p = qbase + (size_t)r * D_ + kk * 32 + hh * 16;
        v8bf lo = *(const v8bf*)p;
        v8bf hi = *(const v8bf*)(p + 8);
#pragma unroll
        for (int j = 0; j < 8; ++j) { qa[kk][j] = lo[j]; qa[kk][8 + j] = hi[j]; }
    }

    float mrow[8], lsum[8];
    v8f oacc[8];
#pragma unroll
    for (int v = 0; v < 8; ++v) { mrow[v] = -1e30f; lsum[v] = 0.f; }
#pragma unroll
    for (int dt = 0; dt < 8; ++dt) oacc[dt] = (v8f){};

    const int lastc = (qt * 16 + 15) >> 5;
    for (int kc = 0; kc <= lastc; ++kc) {
        v8f c0 = {}, c1 = {};
#pragma unroll
        for (int kk = 0; kk < 4; ++kk) {
            const __bf16* kp0 = kbase + (size_t)(kc * 32 + r) * D_ + kk * 32 + hh * 16;
            v16bf b0 = *(const v16bf*)kp0;
            v16bf b1 = *(const v16bf*)(kp0 + 16 * D_);
            c0 = __builtin_amdgcn_wmma_f32_16x16x32_bf16(
                false, qa[kk], false, b0, (short)0, c0, false, false);
            c1 = __builtin_amdgcn_wmma_f32_16x16x32_bf16(
                false, qa[kk], false, b1, (short)0, c1, false, false);
        }
        float p0[8], p1[8], alpha[8];
#pragma unroll
        for (int v = 0; v < 8; ++v) {
            int row  = qt * 16 + v + 8 * hh;
            int col0 = kc * 32 + r;
            float s0 = c0[v] * scale + ((col0      > row) ? NEGM_ : 0.f);
            float s1 = c1[v] * scale + ((col0 + 16 > row) ? NEGM_ : 0.f);
            float mc = fmaxf(s0, s1);
#pragma unroll
            for (int mm = 8; mm >= 1; mm >>= 1) mc = fmaxf(mc, __shfl_xor(mc, mm, 32));
            float mn = fmaxf(mrow[v], mc);
            alpha[v] = __expf(mrow[v] - mn);
            p0[v] = __expf(s0 - mn);
            p1[v] = __expf(s1 - mn);
            float ls = p0[v] + p1[v];
#pragma unroll
            for (int mm = 8; mm >= 1; mm >>= 1) ls += __shfl_xor(ls, mm, 32);
            lsum[v] = lsum[v] * alpha[v] + ls;
            mrow[v] = mn;
        }
#pragma unroll
        for (int dt = 0; dt < 8; ++dt)
#pragma unroll
            for (int v = 0; v < 8; ++v) oacc[dt][v] *= alpha[v];

#pragma unroll
        for (int v = 0; v < 8; ++v) {
            int row = v + 8 * hh;
            pl[row * 32 + r]      = (__bf16)p0[v];
            pl[row * 32 + 16 + r] = (__bf16)p1[v];
        }
        asm volatile("s_wait_dscnt 0" ::: "memory");
        v16bf pa;
        {
            const __bf16* pp = pl + r * 32 + hh * 16;
            v8bf lo = *(const v8bf*)pp;
            v8bf hi = *(const v8bf*)(pp + 8);
#pragma unroll
            for (int j = 0; j < 8; ++j) { pa[j] = lo[j]; pa[8 + j] = hi[j]; }
        }
#pragma unroll
        for (int dt = 0; dt < 8; ++dt) {
            const __bf16* vp = vbase + (size_t)(dt * 16 + r) * S_ + kc * 32 + hh * 16;
            v16bf bv = *(const v16bf*)vp;
            oacc[dt] = __builtin_amdgcn_wmma_f32_16x16x32_bf16(
                false, pa, false, bv, (short)0, oacc[dt], false, false);
        }
    }

    float inv[8];
#pragma unroll
    for (int v = 0; v < 8; ++v) inv[v] = 1.0f / lsum[v];
#pragma unroll
    for (int dt = 0; dt < 8; ++dt)
#pragma unroll
        for (int v = 0; v < 8; ++v) {
            int s = qt * 16 + v + 8 * hh;
            aout[(size_t)(b * S_ + s) * (H_ * D_) + h * D_ + dt * 16 + r] =
                (__bf16)(oacc[dt][v] * inv[v]);
        }
}

// ---------------------------------------------------------------------------
// Host-side launch
// ---------------------------------------------------------------------------
extern "C" void kernel_launch(void* const* d_in, const int* in_sizes, int n_in,
                              void* d_out, int out_size, void* d_ws, size_t ws_size,
                              hipStream_t stream) {
    const float* x    = (const float*)d_in[0];
    const float* cosT = (const float*)d_in[1];
    const float* sinT = (const float*)d_in[2];
    // d_in[3] = mask (unused; causal mask computed analytically)
    const float* wq   = (const float*)d_in[4];
    const float* wk   = (const float*)d_in[5];
    const float* wv   = (const float*)d_in[6];
    const float* wo   = (const float*)d_in[7];
    const float* qnw  = (const float*)d_in[8];
    const float* knw  = (const float*)d_in[9];
    float* out = (float*)d_out;

    const size_t M = (size_t)B_ * S_;          // 4096
    char* p = (char*)d_ws;
    __bf16* xbf   = (__bf16*)p;  p += M * HS_ * 2;
    __bf16* wqkv  = (__bf16*)p;  p += (size_t)NQKV_ * HS_ * 2;
    __bf16* wobf  = (__bf16*)p;  p += (size_t)HS_ * (H_ * D_) * 2;
    float*  qkvf  = (float*)p;   p += M * NQKV_ * 4;
    __bf16* qhb   = (__bf16*)p;  p += (size_t)B_ * H_ * S_ * D_ * 2;
    __bf16* khb   = (__bf16*)p;  p += (size_t)B_ * HKV_ * S_ * D_ * 2;
    __bf16* vtb   = (__bf16*)p;  p += (size_t)B_ * HKV_ * D_ * S_ * 2;
    __bf16* attnb = (__bf16*)p;  p += M * (H_ * D_) * 2;

    // 1) converts
    {
        int n;
        n = (int)(M * HS_);
        cvt_f32_bf16<<<(n + 255) / 256, 256, 0, stream>>>(x, xbf, n);
        n = H_ * D_ * HS_;
        cvt_f32_bf16<<<(n + 255) / 256, 256, 0, stream>>>(wq, wqkv, n);
        int nk = HKV_ * D_ * HS_;
        cvt_f32_bf16<<<(nk + 255) / 256, 256, 0, stream>>>(wk, wqkv + (size_t)H_ * D_ * HS_, nk);
        cvt_f32_bf16<<<(nk + 255) / 256, 256, 0, stream>>>(wv, wqkv + (size_t)(H_ + HKV_) * D_ * HS_, nk);
        n = HS_ * H_ * D_;
        cvt_f32_bf16<<<(n + 255) / 256, 256, 0, stream>>>(wo, wobf, n);
    }

    // 2) fused QKV projection: [4096, 3072] = x @ [wq;wk;wv]^T
    gemm_bf16_f32<<<dim3(NQKV_ / 256, (int)(M / 64)), 256, 0, stream>>>(
        xbf, wqkv, qkvf, (int)M, NQKV_, HS_);

    // 3) RMS QK-norm + RoPE, split heads, transpose V
    qk_norm_rope<<<dim3(H_ + 2 * HKV_, S_, B_), 128, 0, stream>>>(
        qkvf, cosT, sinT, qnw, knw, qhb, khb, vtb);

    // 4) causal flash attention (GQA)
    flash_attn<<<dim3(S_ / 64, H_, B_), 128, 0, stream>>>(qhb, khb, vtb, attnb);

    // 5) output projection straight into d_out (f32)
    gemm_bf16_f32<<<dim3(HS_ / 256, (int)(M / 64)), 256, 0, stream>>>(
        attnb, wobf, out, (int)M, HS_, H_ * D_);

    (void)in_sizes; (void)n_in; (void)out_size; (void)ws_size;
}
